// MultiHeadAttention_12051678233344
// MI455X (gfx1250) — matrix operand order repaired and checked by reading, not compiled, not run on hardware
//
#include <hip/hip_runtime.h>
#include <hip/hip_bf16.h>

// ---------------- problem constants (match reference) ----------------
constexpr int EMBED = 1024;
constexpr int HEADS = 16;
constexpr int HDIM  = 64;
constexpr int BATCH = 2;
constexpr int SEQ   = 2048;
constexpr int MROWS = BATCH * SEQ;          // 4096 GEMM rows
constexpr float SCALE = 0.125f;             // 1/sqrt(64)

// ---------------- CDNA5 WMMA plumbing ----------------
typedef __attribute__((ext_vector_type(16))) __bf16 v16bf;
typedef __attribute__((ext_vector_type(8)))  float  v8f;

__device__ __forceinline__ v8f wmma_bf16(v16bf a, v16bf b, v8f c) {
    // D = A(16x32 bf16) * B(32x16 bf16) + C(16x16 f32)
    return __builtin_amdgcn_wmma_f32_16x16x32_bf16(
        /*neg_a=*/false, a, /*neg_b=*/false, b,
        /*c_mod=*/(short)0, c, /*reuse_a=*/false, /*reuse_b=*/false);
}

// async bulk mover: 16B per lane, global -> LDS, tracked by ASYNCcnt
__device__ __forceinline__ void async_copy_b128(unsigned lds_byte_addr,
                                                const void* gaddr) {
    asm volatile("global_load_async_to_lds_b128 %0, %1, off"
                 :: "v"(lds_byte_addr), "v"((unsigned long long)gaddr)
                 : "memory");
}
__device__ __forceinline__ void wait_asynccnt0() {
    asm volatile("s_wait_asynccnt 0x0" ::: "memory");
}
__device__ __forceinline__ void wait_dscnt0() {
    asm volatile("s_wait_dscnt 0x0" ::: "memory");
}

// =====================================================================
// GEMM: Out[m][n] = sum_k A[m][k] * W[n][k] + bias[n]
//   A_BF16: A already bf16 (context; staged via async-LDS DMA)
//           vs f32 (x; staged with convert)
//   HEADS_OUT: write bf16 into [B,H,S,D] layout; else f32 row-major [M,N]
// Block = 256 threads (8 waves), tile 64x64, K-chunk 32.
// =====================================================================
template <bool A_BF16, bool HEADS_OUT>
__global__ __launch_bounds__(256)
void gemm_xwT(const void* __restrict__ Ap, const float* __restrict__ W,
              const float* __restrict__ bias, void* __restrict__ Outp,
              int M, int N, int K)
{
    __shared__ __bf16 sA[64][40];   // 64 rows x 32 K (padded)
    __shared__ __bf16 sB[64][40];   // 64 W-rows x 32 K (padded)

    const int tid  = threadIdx.x;
    const int w    = tid >> 5;
    const int lane = tid & 31;
    const int m0   = blockIdx.y * 64;
    const int n0   = blockIdx.x * 64;
    const int wm   = (w >> 1) * 16;   // wave row offset in tile
    const int wn   = (w & 1) * 32;    // wave col offset in tile

    const float*  Af = (const float*)Ap;
    const __bf16* Ab = (const __bf16*)Ap;

    v8f c0 = {}; v8f c1 = {};

    for (int k0 = 0; k0 < K; k0 += 32) {
        if (A_BF16) {
            // A tile is already bf16: raw async DMA, one b128 chunk/thread.
            // Guard WAR against last iteration's in-flight LDS reads.
            wait_dscnt0();
            {
                const int r = tid >> 2, part = tid & 3;   // 64 rows x 4 chunks
                const unsigned laddr =
                    (unsigned)(uintptr_t)(&sA[0][0]) + r * (40 * 2) + part * 16;
                async_copy_b128(laddr, Ab + (size_t)(m0 + r) * K + k0 + part * 8);
            }
            for (int i = tid; i < 64 * 32; i += 256) {
                const int r = i >> 5, cc = i & 31;
                sB[r][cc] = (__bf16)W[(size_t)(n0 + r) * K + k0 + cc];
            }
            wait_asynccnt0();
        } else {
            // f32 inputs: stage + convert to bf16 (coalesced loads)
            for (int i = tid; i < 64 * 32; i += 256) {
                const int r = i >> 5, cc = i & 31;
                sA[r][cc] = (__bf16)Af[(size_t)(m0 + r) * K + k0 + cc];
                sB[r][cc] = (__bf16)W[(size_t)(n0 + r) * K + k0 + cc];
            }
            if (k0 + 32 < K) {   // prefetch next K-tile (global_prefetch_b8)
                const int r = tid & 63;
                __builtin_prefetch((const char*)&Af[(size_t)(m0 + r) * K + k0 + 32], 0, 0);
                __builtin_prefetch((const char*)&W[(size_t)(n0 + r) * K + k0 + 32], 0, 0);
            }
        }
        __syncthreads();

        // ---- A fragment: rows wm..wm+15, K 0..31 of staged tile ----
        v16bf a;
        {
            const int mr = wm + (lane & 15);
            const int kb = (lane < 16) ? 0 : 16;  // contiguous half-wave K split (matches B)
            #pragma unroll
            for (int i = 0; i < 8; ++i) {
                a[i]     = sA[mr][kb + i];
                a[8 + i] = sA[mr][kb + 8 + i];
            }
        }
        // ---- B fragments: B[k][n] = W[n][k]; two 16-col subtiles ----
        v16bf b0, b1;
        {
            const int nr0 = wn + (lane & 15);
            const int nr1 = wn + 16 + (lane & 15);
            const int kk  = (lane < 16) ? 0 : 16; // contiguous half-wave K split (matches A)
            #pragma unroll
            for (int i = 0; i < 16; ++i) {
                b0[i] = sB[nr0][kk + i];
                b1[i] = sB[nr1][kk + i];
            }
        }
        c0 = wmma_bf16(a, b0, c0);
        c1 = wmma_bf16(a, b1, c1);
        __syncthreads();
    }

    // ---- epilogue: C layout row = vgpr + 8*(lane>=16), col = lane&15 ----
    const int nl = lane & 15;
    #pragma unroll
    for (int i = 0; i < 8; ++i) {
        const int m  = m0 + wm + i + ((lane >> 4) << 3);
        const int na = n0 + wn + nl;
        const int nb = n0 + wn + 16 + nl;
        const float va = c0[i] + bias[na];
        const float vb = c1[i] + bias[nb];
        if (HEADS_OUT) {
            __bf16* Ob = (__bf16*)Outp;
            const int bb = m / SEQ, s = m % SEQ;
            {
                const int h = na / HDIM, d = na % HDIM;
                Ob[((size_t)(bb * HEADS + h) * SEQ + s) * HDIM + d] = (__bf16)va;
            }
            {
                const int h = nb / HDIM, d = nb % HDIM;
                Ob[((size_t)(bb * HEADS + h) * SEQ + s) * HDIM + d] = (__bf16)vb;
            }
        } else {
            float* Of = (float*)Outp;
            Of[(size_t)m * N + na] = va;
            Of[(size_t)m * N + nb] = vb;
        }
    }
}

// =====================================================================
// Fused attention: one block per (b, h, 16 query rows).
// Pass A: online row max/sumexp (per-lane running stats -> 16-lane
//         butterfly -> cross-wave LDS combine).
// Pass B: async-DMA the contiguous 4KB V tile into LDS (overlapped with
//         the QK^T WMMAs + softmax math), recompute score tiles, write
//         probs/H via f32 global atomics into avg_attn, accumulate P*V
//         with WMMA, reduce partial context across waves.
// =====================================================================
__global__ __launch_bounds__(256)
void attn_fused(const __bf16* __restrict__ Q, const __bf16* __restrict__ Kmat,
                const __bf16* __restrict__ V, __bf16* __restrict__ ctx,
                float* __restrict__ avg)
{
    const int tid  = threadIdx.x;
    const int w    = tid >> 5;
    const int lane = tid & 31;
    const int row0 = blockIdx.x * 16;
    const int h    = blockIdx.y;
    const int b    = blockIdx.z;

    const __bf16* qb = Q    + ((size_t)(b * HEADS + h) * SEQ + row0) * HDIM;
    const __bf16* kb = Kmat + (size_t)(b * HEADS + h) * SEQ * HDIM;
    const __bf16* vb = V    + (size_t)(b * HEADS + h) * SEQ * HDIM;

    __shared__ float  s_mlw[8][16][2];     // per-wave partial (m, l)
    __shared__ float  s_m[16], s_linv[16]; // final rowmax, 1/rowsum
    __shared__ __bf16 s_p[8][16][32];      // per-wave probs tile
    // per-wave 4KB overlay: V staging tile (32x64 bf16) during the key
    // loop, then partial-context buffer (16x64 f32) for the reduction.
    __shared__ char   s_ovl[8][4096];

    // ---- Q A-fragments for d = 0..31 and 32..63, fixed for the block ----
    v16bf aq0, aq1;
    {
        const int m   = lane & 15;
        const int kbo = (lane < 16) ? 0 : 16;
        #pragma unroll
        for (int i = 0; i < 8; ++i) {
            aq0[i]     = qb[m * HDIM + kbo + i];
            aq0[8 + i] = qb[m * HDIM + kbo + 8 + i];
            aq1[i]     = qb[m * HDIM + 32 + kbo + i];
            aq1[8 + i] = qb[m * HDIM + 32 + kbo + 8 + i];
        }
    }

    // scores for 16 keys starting at kt + sub*16 (unscaled)
    auto score_frag = [&](int kt, int sub) -> v8f {
        v16bf bk0, bk1;
        const int key = kt + sub * 16 + (lane & 15);
        const int dko = (lane < 16) ? 0 : 16;
        #pragma unroll
        for (int i = 0; i < 16; ++i) {
            bk0[i] = kb[(size_t)key * HDIM + dko + i];        // B[dk][n]=K[n][dk]
            bk1[i] = kb[(size_t)key * HDIM + 32 + dko + i];
        }
        v8f c = {};
        c = wmma_bf16(aq0, bk0, c);
        c = wmma_bf16(aq1, bk1, c);
        return c;
    };

    // ---------------- pass A: row max / sumexp ----------------
    float mrun[8], lrun[8];
    #pragma unroll
    for (int i = 0; i < 8; ++i) { mrun[i] = -1e30f; lrun[i] = 0.0f; }

    for (int kt = w * 32; kt < SEQ; kt += 256) {
        #pragma unroll
        for (int sub = 0; sub < 2; ++sub) {
            v8f c = score_frag(kt, sub);
            #pragma unroll
            for (int i = 0; i < 8; ++i) {
                const float s  = c[i] * SCALE;
                const float nm = fmaxf(mrun[i], s);
                lrun[i] = lrun[i] * __expf(mrun[i] - nm) + __expf(s - nm);
                mrun[i] = nm;
            }
        }
    }
    // butterfly across the 16 lanes of each half-wave (keys dimension)
    #pragma unroll
    for (int off = 1; off < 16; off <<= 1) {
        #pragma unroll
        for (int i = 0; i < 8; ++i) {
            const float mo = __shfl_xor(mrun[i], off, 32);
            const float lo = __shfl_xor(lrun[i], off, 32);
            const float nm = fmaxf(mrun[i], mo);
            lrun[i] = lrun[i] * __expf(mrun[i] - nm) + lo * __expf(mo - nm);
            mrun[i] = nm;
        }
    }
    if ((lane & 15) == 0) {
        const int rbase = (lane >> 4) << 3;   // 0 or 8
        #pragma unroll
        for (int i = 0; i < 8; ++i) {
            s_mlw[w][rbase + i][0] = mrun[i];
            s_mlw[w][rbase + i][1] = lrun[i];
        }
    }
    __syncthreads();
    if (tid < 16) {
        float m = -1e30f, l = 0.0f;
        #pragma unroll
        for (int ww = 0; ww < 8; ++ww) {
            const float mo = s_mlw[ww][tid][0];
            const float lo = s_mlw[ww][tid][1];
            const float nm = fmaxf(m, mo);
            l = l * __expf(m - nm) + lo * __expf(mo - nm);
            m = nm;
        }
        s_m[tid]    = m;
        s_linv[tid] = 1.0f / l;
    }
    __syncthreads();

    // ---------------- pass B: probs, avg_attn, P*V ----------------
    v8f o0 = {}, o1 = {}, o2 = {}, o3 = {};
    float* avg_base = avg + ((size_t)b * SEQ + row0) * SEQ;
    const unsigned vtile_lds = (unsigned)(uintptr_t)(&s_ovl[w][0]);
    const __bf16*  sV        = (const __bf16*)(&s_ovl[w][0]);   // [32][64]

    for (int kt = w * 32; kt < SEQ; kt += 256) {   // 8 iters for every wave
        // kick off async DMA of this iteration's contiguous 4KB V tile;
        // it overlaps with the score WMMAs / softmax below.
        wait_dscnt0();                 // WAR vs. prior iter's sV ds-reads
        {
            const char* vsrc = (const char*)(vb + (size_t)kt * HDIM);
            #pragma unroll
            for (int j = 0; j < 8; ++j) {
                const int u = j * 32 + lane;       // 256 x 16B chunks
                async_copy_b128(vtile_lds + u * 16, vsrc + u * 16);
            }
        }

        #pragma unroll
        for (int sub = 0; sub < 2; ++sub) {
            v8f c = score_frag(kt, sub);
            const int col = sub * 16 + (lane & 15);
            #pragma unroll
            for (int i = 0; i < 8; ++i) {
                const int   r = i + ((lane >> 4) << 3);
                const float p = __expf(c[i] * SCALE - s_m[r]) * s_linv[r];
                s_p[w][r][col] = (__bf16)p;
                atomicAdd(&avg_base[(size_t)r * SEQ + kt + col],
                          p * (1.0f / (float)HEADS));
            }
        }
        __syncthreads();   // uniform: every wave runs exactly 8 iterations

        // probs A-fragment (16 rows x 32 keys)
        v16bf pa;
        {
            const int m   = lane & 15;
            const int kbo = (lane < 16) ? 0 : 16;
            #pragma unroll
            for (int i = 0; i < 8; ++i) {
                pa[i]     = s_p[w][m][kbo + i];
                pa[8 + i] = s_p[w][m][kbo + 8 + i];
            }
        }

        wait_asynccnt0();              // V tile landed in LDS
        // V B-fragments from LDS: 4 column chunks of 16 over HDIM=64
        #pragma unroll
        for (int j = 0; j < 4; ++j) {
            v16bf bv;
            const int n  = j * 16 + (lane & 15);
            const int t0 = (lane < 16) ? 0 : 16;
            #pragma unroll
            for (int i = 0; i < 16; ++i)
                bv[i] = sV[(t0 + i) * HDIM + n];
            switch (j) {
                case 0:  o0 = wmma_bf16(pa, bv, o0); break;
                case 1:  o1 = wmma_bf16(pa, bv, o1); break;
                case 2:  o2 = wmma_bf16(pa, bv, o2); break;
                default: o3 = wmma_bf16(pa, bv, o3); break;
            }
        }
        __syncthreads();
    }

    // ---- reduce partial context across the 8 waves (overlay as f32) ----
    float* s_o = (float*)(&s_ovl[w][0]);          // [16][64] per wave
    #pragma unroll
    for (int j = 0; j < 4; ++j) {
        const v8f oo = (j == 0) ? o0 : (j == 1) ? o1 : (j == 2) ? o2 : o3;
        #pragma unroll
        for (int i = 0; i < 8; ++i) {
            const int r = i + ((lane >> 4) << 3);
            s_o[r * 64 + j * 16 + (lane & 15)] = oo[i];
        }
    }
    __syncthreads();
    for (int idx = tid; idx < 16 * 64; idx += 256) {
        const int r = idx >> 6, d = idx & 63;
        float acc = 0.0f;
        #pragma unroll
        for (int ww = 0; ww < 8; ++ww)
            acc += ((const float*)(&s_ovl[ww][0]))[r * 64 + d];
        ctx[((size_t)b * SEQ + row0 + r) * EMBED + h * HDIM + d] = (__bf16)acc;
    }
}

// =====================================================================
extern "C" void kernel_launch(void* const* d_in, const int* in_sizes, int n_in,
                              void* d_out, int out_size, void* d_ws, size_t ws_size,
                              hipStream_t stream)
{
    (void)in_sizes; (void)n_in; (void)out_size; (void)ws_size;

    const float* x  = (const float*)d_in[0];
    const float* Wq = (const float*)d_in[1];
    const float* bq = (const float*)d_in[2];
    const float* Wk = (const float*)d_in[3];
    const float* bk = (const float*)d_in[4];
    const float* Wv = (const float*)d_in[5];
    const float* bv = (const float*)d_in[6];
    const float* Wo = (const float*)d_in[7];
    const float* bo = (const float*)d_in[8];

    float* out = (float*)d_out;                                 // [B,S,E]
    float* avg = out + (size_t)BATCH * SEQ * EMBED;             // [B,S,S]

    const size_t headsz = (size_t)BATCH * HEADS * SEQ * HDIM;   // == B*S*E
    __bf16* qws = (__bf16*)d_ws;
    __bf16* kws = qws + headsz;
    __bf16* vws = kws + headsz;
    __bf16* cws = vws + headsz;

    // avg_attn is accumulated with atomics -> zero it first (capturable)
    hipMemsetAsync(avg, 0, (size_t)BATCH * SEQ * SEQ * sizeof(float), stream);

    dim3 blk(256);
    dim3 g1(EMBED / 64, MROWS / 64);
    gemm_xwT<false, true><<<g1, blk, 0, stream>>>(x, Wq, bq, qws, MROWS, EMBED, EMBED);
    gemm_xwT<false, true><<<g1, blk, 0, stream>>>(x, Wk, bk, kws, MROWS, EMBED, EMBED);
    gemm_xwT<false, true><<<g1, blk, 0, stream>>>(x, Wv, bv, vws, MROWS, EMBED, EMBED);

    dim3 ga(SEQ / 16, HEADS, BATCH);
    attn_fused<<<ga, blk, 0, stream>>>(qws, kws, vws, cws, avg);

    gemm_xwT<true, false><<<g1, blk, 0, stream>>>(cws, Wo, bo, out, MROWS, EMBED, EMBED);
}
